// LinearBN1d_76982993814281
// MI455X (gfx1250) — matrix operand order repaired and checked by reading, not compiled, not run on hardware
//
#include <hip/hip_runtime.h>
#include <cstdint>
#include <cstddef>

// ---------------------------------------------------------------------------
// LinearBN1d SNN:  spike_out (N,T,Dout), spike_count (N,Dout)
//   N=512, T=64, D_in=1024, D_out=1024
// Plan:
//   k0: fold BN into weight, split into bf16 hi/lo pair + b_norm/T   -> d_ws
//   k1: convert 0/1 input spikes f32 -> bf16 (exact)                 -> d_ws
//   k2: fused WMMA GEMM (per-sample, all T on-chip) + IF scan
//       A tiles staged with global_load_async_to_lds_b128, all 8 waves
//       issuing, double-buffered, ASYNCcnt-throttled.
// ---------------------------------------------------------------------------

typedef __attribute__((ext_vector_type(16))) __bf16 v16bf;
typedef __attribute__((ext_vector_type(8)))  __bf16 v8bf;
typedef __attribute__((ext_vector_type(4)))  __bf16 v4bf;
typedef __attribute__((ext_vector_type(8)))  float  v8f;
typedef int v4i_ __attribute__((vector_size(16)));   // matches builtin param
typedef __attribute__((address_space(1))) v4i_ gv4i; // global int4
typedef __attribute__((address_space(3))) v4i_ sv4i; // LDS int4

#define D_IN    1024
#define D_OUT   1024
#define T_STEPS 64
#define N_BATCH 512
#define KC      64          // K per staged chunk (2 WMMA k-steps)
#define NCHUNK  (D_IN / KC) // 16

#if defined(__gfx1250__) && __has_builtin(__builtin_amdgcn_global_load_async_to_lds_b128) && __has_builtin(__builtin_amdgcn_s_wait_asynccnt)
#define HAVE_ASYNC 1
#else
#define HAVE_ASYNC 0
#endif

__device__ __forceinline__ v8f wmma_bf16(v16bf a, v16bf b, v8f c) {
  // v_wmma_f32_16x16x32_bf16, f32 accumulate
  return __builtin_amdgcn_wmma_f32_16x16x32_bf16(false, a, false, b, (short)0, c,
                                                 false, false);
}

// --------------------------- k0: weight prep -------------------------------
__global__ __launch_bounds__(256) void prep_w(
    const float* __restrict__ w, const float* __restrict__ bias,
    const float* __restrict__ gamma, const float* __restrict__ beta,
    const float* __restrict__ rmean, const float* __restrict__ rvar,
    __bf16* __restrict__ whi, __bf16* __restrict__ wlo,
    float* __restrict__ bstep) {
  const int o = blockIdx.x;
  const float ratio = gamma[o] / sqrtf(rvar[o]);  // note: no eps (matches ref)
  const int k = threadIdx.x * 4;
  const float4 wv = *(const float4*)(w + (size_t)o * D_IN + k);
  float f[4] = {wv.x * ratio, wv.y * ratio, wv.z * ratio, wv.w * ratio};
  v4bf hv, lv;
#pragma unroll
  for (int i = 0; i < 4; ++i) {
    hv[i] = (__bf16)f[i];
    lv[i] = (__bf16)(f[i] - (float)hv[i]);  // bf16x2 split ~= fp32 precision
  }
  *(v4bf*)(whi + (size_t)o * D_IN + k) = hv;
  *(v4bf*)(wlo + (size_t)o * D_IN + k) = lv;
  if (threadIdx.x == 0)
    bstep[o] = ((bias[o] - rmean[o]) * ratio + beta[o]) * (1.0f / T_STEPS);
}

// --------------------------- k1: input f32 -> bf16 -------------------------
__global__ __launch_bounds__(256) void prep_a(const float* __restrict__ in,
                                              __bf16* __restrict__ out) {
  const size_t idx = ((size_t)blockIdx.x * 256 + threadIdx.x) * 8;
  const float4 a = *(const float4*)(in + idx);
  const float4 b = *(const float4*)(in + idx + 4);
  v8bf r;
  r[0] = (__bf16)a.x; r[1] = (__bf16)a.y; r[2] = (__bf16)a.z; r[3] = (__bf16)a.w;
  r[4] = (__bf16)b.x; r[5] = (__bf16)b.y; r[6] = (__bf16)b.z; r[7] = (__bf16)b.w;
  *(v8bf*)(out + idx) = r;
}

// --------------------- k2: fused WMMA GEMM + IF scan -----------------------
// grid(D_OUT/128, N).  8 waves; wave w -> cols [128*bx + 16w, +16), 4 t-tiles.
// A staging layout: row t = 128B covering k0..63; each 32-k group is a 64B
// block in plain k order [k0-7][k8-15][k16-23][k24-31] so each lane's WMMA-A operand
// is one contiguous 32B LDS read.
__global__ __launch_bounds__(256) void snn_gemm_if(
    const __bf16* __restrict__ abf, const __bf16* __restrict__ whi,
    const __bf16* __restrict__ wlo, const float* __restrict__ bstep,
    float* __restrict__ spike_out, float* __restrict__ spike_cnt) {
  __shared__ __align__(128) char  sA[2][T_STEPS * 128];  // 2 x 8 KB
  __shared__ __align__(128) float sPot[128 * 68];        // padded col-major pot

  const int tid  = threadIdx.x;
  const int wv   = tid >> 5;
  const int lane = tid & 31;
  const int l15  = lane & 15;
  const int h    = (lane >> 4) & 1;
  const int n    = blockIdx.y;
  const int cb   = blockIdx.x * 128;

  const int ob = cb + wv * 16 + l15;  // this lane's B column (weight row)
  const __bf16* bh_row = whi + (size_t)ob * D_IN + h * 16;
  const __bf16* bl_row = wlo + (size_t)ob * D_IN + h * 16;
  const __bf16* arow   = abf + (size_t)n * (T_STEPS * D_IN);

  v8f c0 = {}, c1 = {}, c2 = {}, c3 = {};

  // 512 x 16B chunks per tile; every thread moves 2 (all waves issue async).
  auto stage = [&](int kci, int p) {
#pragma unroll
    for (int i = 0; i < 2; ++i) {
      const int id = wv * 64 + i * 32 + lane;
      const int t = id >> 3, c = id & 7;
      const int g = c >> 2, cc = c & 3;
      const __bf16* gp = arow + t * D_IN + kci * KC + c * 8;
      char* lp = &sA[p][t * 128 + g * 64 + cc * 16];
#if HAVE_ASYNC
      __builtin_amdgcn_global_load_async_to_lds_b128(
          (gv4i*)(uintptr_t)gp, (sv4i*)(uint32_t)(uintptr_t)lp, 0, 0);
#else
      *(v8bf*)lp = *(const v8bf*)gp;
#endif
    }
  };

  auto compute = [&](int kci, int p) {
    const char* ab = sA[p];
#pragma unroll
    for (int s = 0; s < 2; ++s) {  // two K=32 steps per staged chunk
      const v16bf bh = *(const v16bf*)(bh_row + kci * KC + s * 32);
      const v16bf bl = *(const v16bf*)(bl_row + kci * KC + s * 32);
      const v16bf a0 = *(const v16bf*)(ab + (0 * 16 + l15) * 128 + s * 64 + h * 32);
      const v16bf a1 = *(const v16bf*)(ab + (1 * 16 + l15) * 128 + s * 64 + h * 32);
      const v16bf a2 = *(const v16bf*)(ab + (2 * 16 + l15) * 128 + s * 64 + h * 32);
      const v16bf a3 = *(const v16bf*)(ab + (3 * 16 + l15) * 128 + s * 64 + h * 32);
      c0 = wmma_bf16(a0, bh, c0); c0 = wmma_bf16(a0, bl, c0);
      c1 = wmma_bf16(a1, bh, c1); c1 = wmma_bf16(a1, bl, c1);
      c2 = wmma_bf16(a2, bh, c2); c2 = wmma_bf16(a2, bl, c2);
      c3 = wmma_bf16(a3, bh, c3); c3 = wmma_bf16(a3, bl, c3);
    }
  };

#if HAVE_ASYNC
  stage(0, 0);
#pragma unroll 1
  for (int kci = 0; kci < NCHUNK; ++kci) {
    const int p = kci & 1;
    if (kci + 1 < NCHUNK) {
      stage(kci + 1, p ^ 1);                 // prefetch next tile
      __builtin_amdgcn_s_wait_asynccnt(2);   // this wave's chunk-kci loads done
    } else {
      __builtin_amdgcn_s_wait_asynccnt(0);
    }
    __syncthreads();             // tile kci visible to all waves
    compute(kci, p);
    __syncthreads();             // done reading before buffer reuse
  }
#else
#pragma unroll 1
  for (int kci = 0; kci < NCHUNK; ++kci) {
    __syncthreads();
    stage(kci, 0);
    __syncthreads();
    compute(kci, 0);
  }
#endif

  // Spill accumulators: sPot[col][t], pitch 68 floats (bank-conflict pad).
  // C layout: lane half h, VGPR v -> t = m*16 + v + 8h, col = 16w + l15.
  {
    float* pp = sPot + (wv * 16 + l15) * 68 + h * 8;
#pragma unroll
    for (int i = 0; i < 8; ++i) pp[ 0 + i] = c0[i];
#pragma unroll
    for (int i = 0; i < 8; ++i) pp[16 + i] = c1[i];
#pragma unroll
    for (int i = 0; i < 8; ++i) pp[32 + i] = c2[i];
#pragma unroll
    for (int i = 0; i < 8; ++i) pp[48 + i] = c3[i];
  }
  __syncthreads();

  // Integrate-and-fire scan: one thread per column, sequential over T.
  if (tid < 128) {
    const int o = cb + tid;
    const float bs = bstep[o];
    const float* pp = sPot + tid * 68;
    float* so = spike_out + (size_t)n * (T_STEPS * D_OUT) + o;
    float pot = 0.f, cnt = 0.f;
#pragma unroll
    for (int t = 0; t < T_STEPS; ++t) {
      pot += pp[t] + bs;
      const float spk = (pot >= 1.0f) ? 1.0f : 0.0f;
      pot -= spk;
      cnt += spk;
      so[(size_t)t * D_OUT] = spk;
    }
    spike_cnt[(size_t)n * D_OUT + o] = cnt;
  }
}

// ---------------------------------------------------------------------------
extern "C" void kernel_launch(void* const* d_in, const int* in_sizes, int n_in,
                              void* d_out, int out_size, void* d_ws,
                              size_t ws_size, hipStream_t stream) {
  (void)in_sizes; (void)n_in; (void)out_size; (void)ws_size;
  const float* in_st  = (const float*)d_in[0];  // (N,T,D_in) 0/1 spikes
  // d_in[1] (input_features_sc) feeds only the un-returned ANN path: unused.
  const float* weight = (const float*)d_in[2];
  const float* bias   = (const float*)d_in[3];
  const float* gamma  = (const float*)d_in[4];
  const float* beta   = (const float*)d_in[5];
  const float* rmean  = (const float*)d_in[6];
  const float* rvar   = (const float*)d_in[7];

  char* ws = (char*)d_ws;
  __bf16* whi  = (__bf16*)(ws);                // 2 MB
  __bf16* wlo  = (__bf16*)(ws + (1u << 21));   // 2 MB
  float*  bst  = (float*)(ws + (1u << 22));    // 4 KB
  __bf16* abf  = (__bf16*)(ws + (1u << 23));   // 64 MB bf16 spikes

  float* spike_out = (float*)d_out;
  float* spike_cnt = spike_out + (size_t)N_BATCH * T_STEPS * D_OUT;

  prep_w<<<dim3(D_OUT), 256, 0, stream>>>(weight, bias, gamma, beta, rmean,
                                          rvar, whi, wlo, bst);
  prep_a<<<dim3((N_BATCH * T_STEPS * D_IN) / (256 * 8)), 256, 0, stream>>>(
      in_st, abf);
  snn_gemm_if<<<dim3(D_OUT / 128, N_BATCH), 256, 0, stream>>>(
      abf, whi, wlo, bst, spike_out, spike_cnt);
}
